// VOXCPM_MAIN_47296179864184
// MI455X (gfx1250) — matrix operand order repaired and checked by reading, not compiled, not run on hardware
//
#include <hip/hip_runtime.h>
#include <cstddef>

// ---------------------------------------------------------------------------
// Model constants (match reference)
// ---------------------------------------------------------------------------
#define DMODEL   1024
#define NH       16
#define NKV      2
#define HD       64
#define FF       2560
#define IDS      1024
#define HIST     1024
#define CT       256
#define LTOT     16
#define LBASE    12
#define KV_LEN   (HIST + IDS)     // 2048
#define GQA      (NH / NKV)       // 8
#define FSQW     32
#define ROPE_SCALE_C 0.35355339059327379f   // 64^-0.25

typedef __attribute__((ext_vector_type(16))) _Float16 v16h;
typedef __attribute__((ext_vector_type(8)))  float    v8f;
typedef __attribute__((ext_vector_type(4)))  unsigned int u32x4;

union HFrag { v16h h; u32x4 q[2]; };

__device__ __forceinline__ v8f wmma_f16(const HFrag& a, const HFrag& b, v8f c) {
    return __builtin_amdgcn_wmma_f32_16x16x32_f16(
        /*neg_a=*/false, a.h, /*neg_b=*/false, b.h,
        /*c_mod=*/(short)0, c, /*reuse_a=*/false, /*reuse_b=*/false);
}

// XOR-swizzled LDS indexing (units: halfs). Lines of 32 / 64 halfs are split
// into 16-byte chunks; the chunk index is XORed with low line bits so that
// both staging stores and WMMA fragment loads spread across the 64 LDS banks
// while every ds_load_b128 stays 16B-aligned.
__device__ __forceinline__ int sw32(int line, int khalf) {
    return line * 32 + ((((khalf >> 3) ^ (line >> 2)) & 3) << 3) + (khalf & 7);
}
__device__ __forceinline__ int sw64(int line, int khalf) {
    return line * 64 + ((((khalf >> 3) ^ (line >> 1)) & 7) << 3) + (khalf & 7);
}

// Convert 8 consecutive f32 -> 8 f16 and store as one 16B LDS chunk.
__device__ __forceinline__ void cvt8_store(const float* __restrict__ src,
                                           _Float16* __restrict__ dst) {
    float4 f0 = *reinterpret_cast<const float4*>(src);
    float4 f1 = *reinterpret_cast<const float4*>(src + 4);
    union { _Float16 h[8]; u32x4 v; } pk;
    pk.h[0] = (_Float16)f0.x; pk.h[1] = (_Float16)f0.y;
    pk.h[2] = (_Float16)f0.z; pk.h[3] = (_Float16)f0.w;
    pk.h[4] = (_Float16)f1.x; pk.h[5] = (_Float16)f1.y;
    pk.h[6] = (_Float16)f1.z; pk.h[7] = (_Float16)f1.w;
    *reinterpret_cast<u32x4*>(dst) = pk.v;
}

// ---------------------------------------------------------------------------
// Generic GEMM: C[M,N] (+)= A[M,K] @ B[K,N], f32 in/out, f16 WMMA compute.
// Block tile 128x128x32, 256 threads (8 waves), wave tile 32x64 (2x4 WMMA).
// Requires M % 128 == 0, K % 32 == 0; N masked (slow path) when tile ragged.
// ---------------------------------------------------------------------------
#define BM 128
#define BN 128
#define BKK 32

__global__ __launch_bounds__(256)
void vox_gemm(const float* __restrict__ A, const float* __restrict__ B,
              float* __restrict__ C, int M, int N, int K, int accum)
{
    __shared__ _Float16 sA[BM * BKK];     // [m][k], swizzled chunks
    __shared__ _Float16 sB[BN * BKK];     // [n][k] (transposed), swizzled

    const int tid  = threadIdx.x;
    const int lane = tid & 31;
    const int wave = tid >> 5;
    const int wm = wave & 3;              // 4 waves along M
    const int wn = wave >> 2;             // 2 waves along N
    const int m0 = blockIdx.x * BM;
    const int n0 = blockIdx.y * BN;
    const int lane15   = lane & 15;
    const int laneHalf = lane >> 4;
    const bool fullN = (n0 + BN) <= N;    // block-uniform

    v8f acc[2][4];
#pragma unroll
    for (int i = 0; i < 2; ++i)
#pragma unroll
        for (int j = 0; j < 4; ++j)
#pragma unroll
            for (int r = 0; r < 8; ++r) acc[i][j][r] = 0.0f;

    for (int kk = 0; kk < K; kk += BKK) {
        // ---- stage A tile (128x32 f32 -> f16, vectorized) ----
        {
            const int r = tid >> 1;
            const int cbase = (tid & 1) * 16;
            const float* src = A + (size_t)(m0 + r) * K + kk + cbase;
            cvt8_store(src,     &sA[sw32(r, cbase)]);
            cvt8_store(src + 8, &sA[sw32(r, cbase + 8)]);
        }
        // ---- stage B tile (32x128 f32 -> f16, transposed to [n][k]) ----
        if (fullN) {
            // fast path: 4x4 register transpose, vector loads + b64 stores
            const int kg = tid >> 5;      // 0..7  (4 k-rows each)
            const int ng = tid & 31;      // 0..31 (4 n-cols each)
            const float* src = B + (size_t)(kk + kg * 4) * N + n0 + ng * 4;
            float fr[4][4];
#pragma unroll
            for (int dk = 0; dk < 4; ++dk) {
                float4 t = *reinterpret_cast<const float4*>(src + (size_t)dk * N);
                fr[dk][0] = t.x; fr[dk][1] = t.y; fr[dk][2] = t.z; fr[dk][3] = t.w;
            }
#pragma unroll
            for (int dn = 0; dn < 4; ++dn) {
                union { _Float16 h[4]; unsigned long long u; } pk;
                pk.h[0] = (_Float16)fr[0][dn]; pk.h[1] = (_Float16)fr[1][dn];
                pk.h[2] = (_Float16)fr[2][dn]; pk.h[3] = (_Float16)fr[3][dn];
                *reinterpret_cast<unsigned long long*>(
                    &sB[sw32(ng * 4 + dn, kg * 4)]) = pk.u;
            }
        } else {
            // ragged-N path (only the tiny N=32 GEMM takes this)
            const int kr = tid >> 3;
            const int cb = (tid & 7) * 16;
#pragma unroll
            for (int j = 0; j < 16; ++j) {
                int n = n0 + cb + j;
                float v = 0.0f;
                if (n < N) v = B[(size_t)(kk + kr) * N + n];
                sB[sw32(cb + j, kr)] = (_Float16)v;
            }
        }
        // prefetch next K-tile while this one computes
        if (kk + BKK < K) {
            __builtin_prefetch(A + (size_t)(m0 + (tid >> 1)) * K + kk + BKK, 0, 0);
            __builtin_prefetch(B + (size_t)(kk + BKK + (tid >> 3)) * N + n0, 0, 0);
        }
        __syncthreads();

        // ---- fragments + WMMA ----
        HFrag a[2], b[4];
#pragma unroll
        for (int i = 0; i < 2; ++i) {
            int row = wm * 32 + i * 16 + lane15;
            int kc0 = laneHalf * 8;
            a[i].q[0] = *reinterpret_cast<const u32x4*>(&sA[sw32(row, kc0)]);
            a[i].q[1] = *reinterpret_cast<const u32x4*>(&sA[sw32(row, kc0 + 16)]);
        }
#pragma unroll
        for (int j = 0; j < 4; ++j) {
            int col = wn * 64 + j * 16 + lane15;
            int k0  = laneHalf * 8;
            b[j].q[0] = *reinterpret_cast<const u32x4*>(&sB[sw32(col, k0)]);
            b[j].q[1] = *reinterpret_cast<const u32x4*>(&sB[sw32(col, k0 + 16)]);
        }
#pragma unroll
        for (int i = 0; i < 2; ++i)
#pragma unroll
            for (int j = 0; j < 4; ++j)
                acc[i][j] = wmma_f16(a[i], b[j], acc[i][j]);
        __syncthreads();
    }

    // ---- store C (C layout: VGPR r -> row r / r+8, lane -> col) ----
#pragma unroll
    for (int i = 0; i < 2; ++i) {
#pragma unroll
        for (int j = 0; j < 4; ++j) {
            int col = n0 + wn * 64 + j * 16 + lane15;
            if (col < N) {
#pragma unroll
                for (int r = 0; r < 8; ++r) {
                    int row = m0 + wm * 32 + i * 16 + r + laneHalf * 8;
                    size_t idx = (size_t)row * N + col;
                    float v = acc[i][j][r];
                    C[idx] = accum ? (C[idx] + v) : v;
                }
            }
        }
    }
}

// ---------------------------------------------------------------------------
// RMSNorm: one block per row, D=1024
// ---------------------------------------------------------------------------
__global__ __launch_bounds__(256)
void vox_rmsnorm(const float* __restrict__ x, const float* __restrict__ w,
                 float* __restrict__ out)
{
    __shared__ float red[8];
    __shared__ float s_inv;
    const int row = blockIdx.x;
    const float* xr = x + (size_t)row * DMODEL;
    float s = 0.0f;
    for (int i = threadIdx.x; i < DMODEL; i += 256) { float v = xr[i]; s += v * v; }
#pragma unroll
    for (int off = 16; off; off >>= 1) s += __shfl_xor(s, off, 32);
    if ((threadIdx.x & 31) == 0) red[threadIdx.x >> 5] = s;
    __syncthreads();
    if (threadIdx.x == 0) {
        float t = 0.0f;
#pragma unroll
        for (int i = 0; i < 8; ++i) t += red[i];
        s_inv = rsqrtf(t / (float)DMODEL + 1e-5f);
    }
    __syncthreads();
    const float inv = s_inv;
    for (int i = threadIdx.x; i < DMODEL; i += 256)
        out[(size_t)row * DMODEL + i] = xr[i] * inv * w[i];
}

// ---------------------------------------------------------------------------
// Copy old KV cache into the k_full / v_full output regions
// ---------------------------------------------------------------------------
__global__ void vox_copy_cache(const float* __restrict__ kc, const float* __restrict__ vc,
                               float* __restrict__ outk, float* __restrict__ outv)
{
    const size_t n = (size_t)LTOT * NKV * HD * HIST;
    for (size_t i = (size_t)blockIdx.x * blockDim.x + threadIdx.x; i < n;
         i += (size_t)gridDim.x * blockDim.x) {
        size_t c = i % HIST; size_t rest = i / HIST;      // rest over (L*NKV*HD)
        outk[rest * KV_LEN + c] = kc[i];
        size_t d = i % HD; size_t r2 = i / HD;
        size_t t = r2 % HIST; size_t lh = r2 / HIST;      // lh over (L*NKV)
        outv[(lh * KV_LEN + t) * HD + d] = vc[i];
    }
}

// ---------------------------------------------------------------------------
// RoPE on q (in place) and k, scatter rotated k and v into k_full / v_full.
// ---------------------------------------------------------------------------
__global__ __launch_bounds__(256)
void vox_rope_scatter(float* __restrict__ q, const float* __restrict__ kbuf,
                      const float* __restrict__ vbuf,
                      float* __restrict__ kfull, float* __restrict__ vfull)
{
    const int i   = blockIdx.x;
    const int tid = threadIdx.x;
    const float pos = (float)(HIST + i);

    for (int p = tid; p < NH * 32; p += 256) {
        int h = p >> 5, dp = p & 31;
        float inv = __powf(10000.0f, -(float)(2 * dp) / 64.0f);
        float f = pos * inv;
        float cs = __cosf(f) * ROPE_SCALE_C, sn = __sinf(f) * ROPE_SCALE_C;
        size_t base = (size_t)i * (NH * HD) + (size_t)h * HD;
        float x1 = q[base + dp], x2 = q[base + dp + 32];
        q[base + dp]      = x1 * cs - x2 * sn;
        q[base + dp + 32] = x2 * cs + x1 * sn;
    }
    if (tid < NKV * 32) {
        int h = tid >> 5, dp = tid & 31;
        float inv = __powf(10000.0f, -(float)(2 * dp) / 64.0f);
        float f = pos * inv;
        float cs = __cosf(f) * ROPE_SCALE_C, sn = __sinf(f) * ROPE_SCALE_C;
        size_t base = (size_t)i * (NKV * HD) + (size_t)h * HD;
        float x1 = kbuf[base + dp], x2 = kbuf[base + dp + 32];
        size_t kb = (size_t)h * HD * KV_LEN;
        kfull[kb + (size_t)dp * KV_LEN + HIST + i]        = x1 * cs - x2 * sn;
        kfull[kb + (size_t)(dp + 32) * KV_LEN + HIST + i] = x2 * cs + x1 * sn;
    }
    if (tid < NKV * HD) {
        int h = tid >> 6, d = tid & 63;
        vfull[((size_t)h * KV_LEN + HIST + i) * HD + d] =
            vbuf[(size_t)i * (NKV * HD) + (size_t)h * HD + d];
    }
}

// ---------------------------------------------------------------------------
// Flash-style attention, WMMA for both Q*K^T and P*V.
// grid = (NH, IDS/64), block = 128 (4 waves). Wave owns 16 query rows.
// kfull: (HD, KV_LEN) per kv head; vfull: (KV_LEN, HD) per kv head.
// ---------------------------------------------------------------------------
__global__ __launch_bounds__(128)
void vox_attention(const float* __restrict__ q, const float* __restrict__ kfull,
                   const float* __restrict__ vfull, float* __restrict__ out,
                   const float* __restrict__ mask_flag)
{
    const int head = blockIdx.x;
    const int qb   = blockIdx.y;
    const int kvh  = head / GQA;
    const float* Kp = kfull + (size_t)kvh * HD * KV_LEN;
    const float* Vp = vfull + (size_t)kvh * KV_LEN * HD;

    __shared__ _Float16 sQ[64 * 64];        // [row][d], swizzled
    __shared__ _Float16 sK[64 * 64];        // [kv][d], swizzled
    __shared__ _Float16 sV[64 * 64];        // [d][kv], swizzled
    __shared__ _Float16 sP[4][16 * 64];     // per-wave P tile [row16][kv64]

    const int tid  = threadIdx.x;
    const int lane = tid & 31;
    const int wave = tid >> 5;
    const int lane15   = lane & 15;
    const int laneHalf = lane >> 4;
    const float mf = mask_flag[0];

    // stage Q once (vectorized, contiguous along d)
#pragma unroll
    for (int c = 0; c < 4; ++c) {
        int lin = (c * 128 + tid) * 8;      // multiple of 8
        int r = lin >> 6, d = lin & 63;
        cvt8_store(&q[(size_t)(qb * 64 + r) * (NH * HD) + (size_t)head * HD + d],
                   &sQ[sw64(r, d)]);
    }

    v8f accO[4];
#pragma unroll
    for (int t = 0; t < 4; ++t)
#pragma unroll
        for (int r = 0; r < 8; ++r) accO[t][r] = 0.0f;
    float mrow[8], lrow[8];
#pragma unroll
    for (int r = 0; r < 8; ++r) { mrow[r] = -1e30f; lrow[r] = 0.0f; }

    for (int j0 = 0; j0 < KV_LEN; j0 += 64) {
        __syncthreads();   // previous iteration done with sK/sV (and sQ ready)
        // ---- stage K tile: (d, kv) -> sK[kv][d], 4x4 register transpose ----
        for (int u = tid; u < 256; u += 128) {
            int dg = u >> 4, jg = u & 15;
            float fr[4][4];
#pragma unroll
            for (int dd = 0; dd < 4; ++dd) {
                float4 t = *reinterpret_cast<const float4*>(
                    &Kp[(size_t)(dg * 4 + dd) * KV_LEN + j0 + jg * 4]);
                fr[dd][0] = t.x; fr[dd][1] = t.y; fr[dd][2] = t.z; fr[dd][3] = t.w;
            }
#pragma unroll
            for (int dj = 0; dj < 4; ++dj) {
                union { _Float16 h[4]; unsigned long long u64; } pk;
                pk.h[0] = (_Float16)fr[0][dj]; pk.h[1] = (_Float16)fr[1][dj];
                pk.h[2] = (_Float16)fr[2][dj]; pk.h[3] = (_Float16)fr[3][dj];
                *reinterpret_cast<unsigned long long*>(
                    &sK[sw64(jg * 4 + dj, dg * 4)]) = pk.u64;
            }
        }
        // ---- stage V tile: (kv, d) -> sV[d][kv], 4x4 register transpose ----
        for (int u = tid; u < 256; u += 128) {
            int jg = u >> 4, dg = u & 15;
            float fr[4][4];
#pragma unroll
            for (int dj = 0; dj < 4; ++dj) {
                float4 t = *reinterpret_cast<const float4*>(
                    &Vp[(size_t)(j0 + jg * 4 + dj) * HD + dg * 4]);
                fr[dj][0] = t.x; fr[dj][1] = t.y; fr[dj][2] = t.z; fr[dj][3] = t.w;
            }
#pragma unroll
            for (int dd = 0; dd < 4; ++dd) {
                union { _Float16 h[4]; unsigned long long u64; } pk;
                pk.h[0] = (_Float16)fr[0][dd]; pk.h[1] = (_Float16)fr[1][dd];
                pk.h[2] = (_Float16)fr[2][dd]; pk.h[3] = (_Float16)fr[3][dd];
                *reinterpret_cast<unsigned long long*>(
                    &sV[sw64(dg * 4 + dd, jg * 4)]) = pk.u64;
            }
        }
        if (j0 + 64 < KV_LEN) {
            __builtin_prefetch(&Kp[(size_t)(tid & 63) * KV_LEN + j0 + 64], 0, 0);
            __builtin_prefetch(&Vp[(size_t)(j0 + 64 + (tid & 63)) * HD], 0, 0);
        }
        __syncthreads();

        // ---- S = Q(16x64) @ K^T(64x64) ----
        v8f accS[4];
#pragma unroll
        for (int t = 0; t < 4; ++t)
#pragma unroll
            for (int r = 0; r < 8; ++r) accS[t][r] = 0.0f;
#pragma unroll
        for (int kc = 0; kc < 64; kc += 32) {
            HFrag a;
            int row = wave * 16 + lane15;
            int ka  = kc + laneHalf * 8;
            a.q[0] = *reinterpret_cast<const u32x4*>(&sQ[sw64(row, ka)]);
            a.q[1] = *reinterpret_cast<const u32x4*>(&sQ[sw64(row, ka + 16)]);
#pragma unroll
            for (int t = 0; t < 4; ++t) {
                HFrag b;
                int col = t * 16 + lane15;
                int kb  = kc + laneHalf * 8;
                b.q[0] = *reinterpret_cast<const u32x4*>(&sK[sw64(col, kb)]);
                b.q[1] = *reinterpret_cast<const u32x4*>(&sK[sw64(col, kb + 16)]);
                accS[t] = wmma_f16(a, b, accS[t]);
            }
        }

        // ---- mask + online softmax (rows live in 16-lane groups) ----
        float scale[8];
#pragma unroll
        for (int r = 0; r < 8; ++r) {
            int grow = qb * 64 + wave * 16 + r + laneHalf * 8;
            float vmax = -1e30f;
#pragma unroll
            for (int t = 0; t < 4; ++t) {
                float s = accS[t][r];
                int col = j0 + t * 16 + lane15;
                if (col > grow) s += -128.0f * mf;
                accS[t][r] = s;
                vmax = fmaxf(vmax, s);
            }
#pragma unroll
            for (int off = 8; off; off >>= 1)
                vmax = fmaxf(vmax, __shfl_xor(vmax, off, 32));
            float mnew = fmaxf(mrow[r], vmax);
            float sc = __expf(mrow[r] - mnew);
            float ps = 0.0f;
#pragma unroll
            for (int t = 0; t < 4; ++t) {
                float p = __expf(accS[t][r] - mnew);
                sP[wave][(r + laneHalf * 8) * 64 + t * 16 + lane15] = (_Float16)p;
                ps += p;
            }
#pragma unroll
            for (int off = 8; off; off >>= 1) ps += __shfl_xor(ps, off, 32);
            lrow[r] = lrow[r] * sc + ps;
            mrow[r] = mnew;
            scale[r] = sc;
        }
#pragma unroll
        for (int t = 0; t < 4; ++t)
#pragma unroll
            for (int r = 0; r < 8; ++r) accO[t][r] *= scale[r];
        __syncthreads();

        // ---- O += P(16x64) @ V(64x64) ----
#pragma unroll
        for (int kc = 0; kc < 64; kc += 32) {
            HFrag a;
            int row = lane15;
            int ka  = kc + laneHalf * 8;
            a.q[0] = *reinterpret_cast<const u32x4*>(&sP[wave][row * 64 + ka]);
            a.q[1] = *reinterpret_cast<const u32x4*>(&sP[wave][row * 64 + ka + 16]);
#pragma unroll
            for (int t = 0; t < 4; ++t) {
                HFrag b;
                int col = t * 16 + lane15;
                int kb  = kc + laneHalf * 8;
                b.q[0] = *reinterpret_cast<const u32x4*>(&sV[sw64(col, kb)]);
                b.q[1] = *reinterpret_cast<const u32x4*>(&sV[sw64(col, kb + 16)]);
                accO[t] = wmma_f16(a, b, accO[t]);
            }
        }
    }

    // ---- write O / l ----
#pragma unroll
    for (int t = 0; t < 4; ++t) {
        int col = head * HD + t * 16 + lane15;
#pragma unroll
        for (int r = 0; r < 8; ++r) {
            int row = qb * 64 + wave * 16 + r + laneHalf * 8;
            out[(size_t)row * (NH * HD) + col] = accO[t][r] / lrow[r];
        }
    }
}

// ---------------------------------------------------------------------------
// Elementwise kernels
// ---------------------------------------------------------------------------
__global__ void vox_silu_mul(const float* __restrict__ g, const float* __restrict__ u,
                             float* __restrict__ out, size_t n)
{
    for (size_t i = (size_t)blockIdx.x * blockDim.x + threadIdx.x; i < n;
         i += (size_t)gridDim.x * blockDim.x) {
        float x = g[i];
        out[i] = (x / (1.0f + __expf(-x))) * u[i];
    }
}

__global__ void vox_fsq_quant(float* __restrict__ z, size_t n)
{
    for (size_t i = (size_t)blockIdx.x * blockDim.x + threadIdx.x; i < n;
         i += (size_t)gridDim.x * blockDim.x) {
        float t = tanhf(z[i]);
        z[i] = roundf(t * 4.0f) * 0.25f;   // straight-through value
    }
}

__global__ void vox_assemble_head(const float* __restrict__ hn,
                                  const float* __restrict__ fsqo,
                                  const float* __restrict__ feat,
                                  float* __restrict__ hnew,
                                  float* __restrict__ lm_hidden)
{
    const size_t n = (size_t)IDS * DMODEL;
    for (size_t i = (size_t)blockIdx.x * blockDim.x + threadIdx.x; i < n;
         i += (size_t)gridDim.x * blockDim.x) {
        size_t r = i / DMODEL, c = i % DMODEL;
        if (r < CT) hnew[i] = hn[i];
        else        hnew[i] = fsqo[(r - CT) * DMODEL + c] + feat[(r - CT) * DMODEL + c];
        if (i < DMODEL)
            lm_hidden[i] = fsqo[(size_t)(IDS - CT - 1) * DMODEL + i];
    }
}

__global__ void vox_dit(const float* __restrict__ lm, const float* __restrict__ res,
                        const float* __restrict__ W1, const float* __restrict__ W2,
                        float* __restrict__ out)
{
    int c = blockIdx.x * blockDim.x + threadIdx.x;
    if (c >= 1024) return;
    float s = 0.0f;
    for (int k = 0; k < DMODEL; ++k)
        s += lm[k] * W1[(size_t)k * 1024 + c] + res[k] * W2[(size_t)k * 1024 + c];
    out[c] = s;
}

__global__ void vox_stop_proj(const float* __restrict__ lm, const float* __restrict__ W,
                              float* __restrict__ s1)
{
    int c = blockIdx.x * blockDim.x + threadIdx.x;
    if (c >= DMODEL) return;
    float s = 0.0f;
    for (int k = 0; k < DMODEL; ++k) s += lm[k] * W[(size_t)k * DMODEL + c];
    float g = 0.5f * s * (1.0f + tanhf(0.7978845608f * (s + 0.044715f * s * s * s)));
    s1[c] = g;
}

__global__ void vox_stop_head(const float* __restrict__ s1, const float* __restrict__ W,
                              float* __restrict__ flag)
{
    if (blockIdx.x == 0 && threadIdx.x == 0) {
        float a = 0.0f, b = 0.0f;
        for (int k = 0; k < DMODEL; ++k) { a += s1[k] * W[k * 2]; b += s1[k] * W[k * 2 + 1]; }
        flag[0] = (b > a) ? 1.0f : 0.0f;
    }
}

__global__ void vox_fill_zero(float* __restrict__ p, int n)
{
    int i = blockIdx.x * blockDim.x + threadIdx.x;
    if (i < n) p[i] = 0.0f;
}

// ---------------------------------------------------------------------------
// Host-side orchestration
// ---------------------------------------------------------------------------
extern "C" void kernel_launch(void* const* d_in, const int* in_sizes, int n_in,
                              void* d_out, int out_size, void* d_ws, size_t ws_size,
                              hipStream_t stream)
{
    (void)in_sizes; (void)n_in; (void)out_size; (void)ws_size;

    const float* hidden    = (const float*)d_in[0];
    const float* feat      = (const float*)d_in[1];
    const float* kcache    = (const float*)d_in[2];
    const float* vcache    = (const float*)d_in[3];
    const float* maskf     = (const float*)d_in[4];
    const float* ln1       = (const float*)d_in[5];
    const float* ln2       = (const float*)d_in[6];
    const float* qw        = (const float*)d_in[7];
    const float* kw        = (const float*)d_in[8];
    const float* vw        = (const float*)d_in[9];
    const float* ow        = (const float*)d_in[10];
    const float* gw        = (const float*)d_in[11];
    const float* uw        = (const float*)d_in[12];
    const float* dw        = (const float*)d_in[13];
    const float* base_norm = (const float*)d_in[14];
    const float* res_norm  = (const float*)d_in[15];
    const float* fsq_down  = (const float*)d_in[16];
    const float* fsq_up    = (const float*)d_in[17];
    const float* lm2dit    = (const float*)d_in[18];
    const float* res2dit   = (const float*)d_in[19];
    const float* stop_pw   = (const float*)d_in[20];
    const float* stop_hw   = (const float*)d_in[21];

    float* out  = (float*)d_out;
    float* newk = out;                                        // (16,2,64,2048)
    float* newv = newk + (size_t)LTOT * NKV * HD * KV_LEN;    // (16,2,2048,64)
    float* dit  = newv + (size_t)LTOT * NKV * KV_LEN * HD;    // (1,1,1024)
    float* rnd  = dit + 1024;                                 // (1,2,64)
    float* flag = rnd + 2 * 64;                               // (1,)

    float* ws   = (float*)d_ws;
    float* h    = ws;
    float* xn   = h    + (size_t)IDS * DMODEL;
    float* qb   = xn   + (size_t)IDS * DMODEL;
    float* ao   = qb   + (size_t)IDS * DMODEL;
    float* kb   = ao   + (size_t)IDS * DMODEL;
    float* vb   = kb   + (size_t)IDS * NKV * HD;
    float* gatb = vb   + (size_t)IDS * NKV * HD;
    float* upb  = gatb + (size_t)IDS * FF;
    float* hn   = upb  + (size_t)IDS * FF;
    float* zb   = hn   + (size_t)IDS * DMODEL;
    float* fsqo = zb   + (size_t)(IDS - CT) * FSQW;
    float* lmh  = fsqo + (size_t)(IDS - CT) * DMODEL;
    float* resh = lmh  + DMODEL;
    float* s1   = resh + DMODEL;

    hipMemcpyAsync(h, hidden, (size_t)IDS * DMODEL * sizeof(float),
                   hipMemcpyDeviceToDevice, stream);
    vox_copy_cache<<<4096, 256, 0, stream>>>(kcache, vcache, newk, newv);

    auto run_layer = [&](int l) {
        const size_t lqw = (size_t)l * DMODEL * (NH * HD);
        const size_t lkw = (size_t)l * DMODEL * (NKV * HD);
        const size_t low = (size_t)l * (NH * HD) * DMODEL;
        const size_t lgw = (size_t)l * DMODEL * FF;
        const size_t ldw = (size_t)l * FF * DMODEL;
        float* kf = newk + (size_t)l * NKV * HD * KV_LEN;
        float* vf = newv + (size_t)l * NKV * KV_LEN * HD;

        vox_rmsnorm<<<IDS, 256, 0, stream>>>(h, ln1 + (size_t)l * DMODEL, xn);
        vox_gemm<<<dim3(IDS / BM, (NH * HD) / BN), 256, 0, stream>>>(
            xn, qw + lqw, qb, IDS, NH * HD, DMODEL, 0);
        vox_gemm<<<dim3(IDS / BM, 1), 256, 0, stream>>>(
            xn, kw + lkw, kb, IDS, NKV * HD, DMODEL, 0);
        vox_gemm<<<dim3(IDS / BM, 1), 256, 0, stream>>>(
            xn, vw + lkw, vb, IDS, NKV * HD, DMODEL, 0);
        vox_rope_scatter<<<IDS, 256, 0, stream>>>(qb, kb, vb, kf, vf);
        vox_attention<<<dim3(NH, IDS / 64), 128, 0, stream>>>(qb, kf, vf, ao, maskf);
        vox_gemm<<<dim3(IDS / BM, DMODEL / BN), 256, 0, stream>>>(
            ao, ow + low, h, IDS, DMODEL, NH * HD, 1);            // h += attn @ o_w
        vox_rmsnorm<<<IDS, 256, 0, stream>>>(h, ln2 + (size_t)l * DMODEL, xn);
        vox_gemm<<<dim3(IDS / BM, FF / BN), 256, 0, stream>>>(
            xn, gw + lgw, gatb, IDS, FF, DMODEL, 0);
        vox_gemm<<<dim3(IDS / BM, FF / BN), 256, 0, stream>>>(
            xn, uw + lgw, upb, IDS, FF, DMODEL, 0);
        vox_silu_mul<<<4096, 256, 0, stream>>>(gatb, upb, gatb, (size_t)IDS * FF);
        vox_gemm<<<dim3(IDS / BM, DMODEL / BN), 256, 0, stream>>>(
            gatb, dw + ldw, h, IDS, DMODEL, FF, 1);               // h += act @ down_w
    };

    for (int l = 0; l < LBASE; ++l) run_layer(l);

    // ---- head between stacks ----
    vox_rmsnorm<<<IDS, 256, 0, stream>>>(h, base_norm, hn);
    vox_gemm<<<dim3((IDS - CT) / BM, 1), 256, 0, stream>>>(
        hn + (size_t)CT * DMODEL, fsq_down, zb, IDS - CT, FSQW, DMODEL, 0);
    vox_fsq_quant<<<96, 256, 0, stream>>>(zb, (size_t)(IDS - CT) * FSQW);
    vox_gemm<<<dim3((IDS - CT) / BM, DMODEL / BN), 256, 0, stream>>>(
        zb, fsq_up, fsqo, IDS - CT, DMODEL, FSQW, 0);
    vox_assemble_head<<<2048, 256, 0, stream>>>(hn, fsqo, feat, h, lmh);

    for (int l = LBASE; l < LTOT; ++l) run_layer(l);

    // ---- final heads ----
    vox_rmsnorm<<<1, 256, 0, stream>>>(h + (size_t)(IDS - 1) * DMODEL, res_norm, resh);
    vox_dit<<<4, 256, 0, stream>>>(lmh, resh, lm2dit, res2dit, dit);
    vox_stop_proj<<<4, 256, 0, stream>>>(lmh, stop_pw, s1);
    vox_fill_zero<<<1, 129, 0, stream>>>(rnd, 129);   // rnd(128) + flag slot init
    vox_stop_head<<<1, 32, 0, stream>>>(s1, stop_hw, flag);
}
